// SGDAT_65446711657153
// MI455X (gfx1250) — matrix operand order repaired and checked by reading, not compiled, not run on hardware
//
#include <hip/hip_runtime.h>
#include <hip/hip_bf16.h>

typedef _Float16 half8  __attribute__((ext_vector_type(8)));
typedef _Float16 half16 __attribute__((ext_vector_type(16)));
typedef float    f32x8  __attribute__((ext_vector_type(8)));

#define NPTS 8192
#define CIN  9

// ---------------- normalize_xyz: center by mean, scale by max radius ----------------
__global__ __launch_bounds__(256) void normalize_k(const float* __restrict__ points,
                                                   float* __restrict__ xyzn) {
  const int b = blockIdx.x, tid = threadIdx.x;
  __shared__ float red[256];
  __shared__ float mean[3];
  __shared__ float scale;
  for (int comp = 0; comp < 3; ++comp) {
    float s = 0.f;
    for (int i = tid; i < NPTS; i += 256)
      s += points[((size_t)b * NPTS + i) * CIN + comp];
    red[tid] = s; __syncthreads();
    for (int st = 128; st > 0; st >>= 1) { if (tid < st) red[tid] += red[tid + st]; __syncthreads(); }
    if (tid == 0) mean[comp] = red[0] / (float)NPTS;
    __syncthreads();
  }
  float mx = 0.f;
  for (int i = tid; i < NPTS; i += 256) {
    const float* p = &points[((size_t)b * NPTS + i) * CIN];
    float dx = p[0] - mean[0], dy = p[1] - mean[1], dz = p[2] - mean[2];
    mx = fmaxf(mx, dx * dx + dy * dy + dz * dz);
  }
  red[tid] = mx; __syncthreads();
  for (int st = 128; st > 0; st >>= 1) { if (tid < st) red[tid] = fmaxf(red[tid], red[tid + st]); __syncthreads(); }
  if (tid == 0) scale = sqrtf(red[0]) + 1e-6f;
  __syncthreads();
  const float inv = 1.f / scale;
  for (int i = tid; i < NPTS; i += 256) {
    const float* p = &points[((size_t)b * NPTS + i) * CIN];
    float* o = &xyzn[((size_t)b * NPTS + i) * 3];
    o[0] = (p[0] - mean[0]) * inv;
    o[1] = (p[1] - mean[1]) * inv;
    o[2] = (p[2] - mean[2]) * inv;
  }
}

// ---------------- farthest point sampling: register-resident points & distances ----------------
// One persistent block per batch. Coordinates of the argmax winner ride through the
// shuffle reduction, so each of the `npoint` serial rounds is pure VALU + 2 barriers.
__global__ __launch_bounds__(1024) void fps_k(const float* __restrict__ xyz,
                                              int* __restrict__ idx_out, int N, int npoint) {
  const int b = blockIdx.x, tid = threadIdx.x;
  const int lane = tid & 31, wave = tid >> 5;
  __shared__ float wv[32], wxs[32], wys[32], wzs[32];
  __shared__ int   wi[32];
  __shared__ float cxyz[3];
  __shared__ int   cur;
  const float* base = xyz + (size_t)b * N * 3;
  float px[8], py[8], pz[8], dst[8];
  #pragma unroll
  for (int k = 0; k < 8; ++k) {
    int i = tid + k * 1024;
    if (i < N) {
      px[k] = base[i * 3 + 0]; py[k] = base[i * 3 + 1]; pz[k] = base[i * 3 + 2];
      dst[k] = 1e10f;
    } else { px[k] = 0.f; py[k] = 0.f; pz[k] = 0.f; dst[k] = -1.f; }
  }
  if (tid == 0) { cur = 0; cxyz[0] = base[0]; cxyz[1] = base[1]; cxyz[2] = base[2]; }
  __syncthreads();
  for (int it = 0; it < npoint; ++it) {
    const int far = cur;                                   // emit BEFORE update (scan semantics)
    const float cx = cxyz[0], cy = cxyz[1], cz = cxyz[2];
    if (tid == 0) idx_out[b * npoint + it] = far;
    float bestv = -1.f; int besti = 1 << 30; float bx = 0.f, by = 0.f, bz = 0.f;
    #pragma unroll
    for (int k = 0; k < 8; ++k) {
      int i = tid + k * 1024;
      if (i < N) {
        float dx = px[k] - cx, dy = py[k] - cy, dz = pz[k] - cz;
        float d  = dx * dx + dy * dy + dz * dz;
        float nd = fminf(dst[k], d);
        dst[k] = nd;
        if (nd > bestv || (nd == bestv && i < besti)) {
          bestv = nd; besti = i; bx = px[k]; by = py[k]; bz = pz[k];
        }
      }
    }
    for (int off = 16; off > 0; off >>= 1) {               // wave32 argmax, first-index ties
      float ov = __shfl_down(bestv, off, 32);
      int   oi = __shfl_down(besti, off, 32);
      float ox = __shfl_down(bx, off, 32);
      float oy = __shfl_down(by, off, 32);
      float oz = __shfl_down(bz, off, 32);
      if (ov > bestv || (ov == bestv && oi < besti)) { bestv = ov; besti = oi; bx = ox; by = oy; bz = oz; }
    }
    if (lane == 0) { wv[wave] = bestv; wi[wave] = besti; wxs[wave] = bx; wys[wave] = by; wzs[wave] = bz; }
    __syncthreads();
    if (tid < 32) {
      bestv = wv[tid]; besti = wi[tid]; bx = wxs[tid]; by = wys[tid]; bz = wzs[tid];
      for (int off = 16; off > 0; off >>= 1) {
        float ov = __shfl_down(bestv, off, 32);
        int   oi = __shfl_down(besti, off, 32);
        float ox = __shfl_down(bx, off, 32);
        float oy = __shfl_down(by, off, 32);
        float oz = __shfl_down(bz, off, 32);
        if (ov > bestv || (ov == bestv && oi < besti)) { bestv = ov; besti = oi; bx = ox; by = oy; bz = oz; }
      }
      if (tid == 0) { cur = besti; cxyz[0] = bx; cxyz[1] = by; cxyz[2] = bz; }
    }
    __syncthreads();
  }
}

// ---------------- zero-fill (uint32 granularity) ----------------
__global__ void fill0_k(unsigned int* __restrict__ p, int total) {
  int i = blockIdx.x * 256 + threadIdx.x;
  if (i < total) p[i] = 0u;
}

// ---------------- pack f32 [Rsrc][C] -> f16 [Rdst][Cs], zero-padded ----------------
__global__ void pack_f16_k(_Float16* __restrict__ dst, const float* __restrict__ src,
                           int Rsrc, int C, int Cs, int total /* Rdst*Cs */) {
  int i = blockIdx.x * 256 + threadIdx.x;
  if (i >= total) return;
  int r = i / Cs, c = i % Cs;
  float v = (r < Rsrc && c < C) ? src[(size_t)r * C + c] : 0.f;
  dst[i] = (_Float16)v;
}

// ---------------- gather rows (f32 contiguous out) ----------------
__global__ void gather_f32_k(float* __restrict__ dst, const float* __restrict__ src,
                             const int* __restrict__ idx, int Nsrc, int np, int C, int total) {
  int i = blockIdx.x * 256 + threadIdx.x;
  if (i >= total) return;
  int c = i % C, ri = i / C, b = ri / np;
  int j = idx[ri];
  dst[i] = src[((size_t)b * Nsrc + j) * C + c];
}

// ---------------- gather rows -> strided f16 slice of a padded concat buffer ----------------
__global__ void gather_f16s_k(_Float16* __restrict__ dst, int Ds, int colOff,
                              const float* __restrict__ src, const int* __restrict__ idx,
                              int Nsrc, int np, int C, int total /* B*np*C */) {
  int i = blockIdx.x * 256 + threadIdx.x;
  if (i >= total) return;
  int c = i % C, ri = i / C, b = ri / np;
  int j = idx[ri];
  dst[(size_t)ri * Ds + colOff + c] = (_Float16)src[((size_t)b * Nsrc + j) * C + c];
}

// ---------------- nearest-neighbor interpolate -> strided f16 slice ----------------
__global__ __launch_bounds__(256) void nn_interp_k(const float* __restrict__ txyz,
                                                   const float* __restrict__ sxyz,
                                                   const float* __restrict__ sfeat,
                                                   _Float16* __restrict__ outh,
                                                   int T, int S, int F, int Ds, int colOff) {
  __shared__ float sx[512 * 3];
  const int t = blockIdx.x * 256 + threadIdx.x;   // global target over B*T (T % 256 == 0)
  const int b = t / T;
  for (int i = threadIdx.x; i < S * 3; i += 256) sx[i] = sxyz[(size_t)b * S * 3 + i];
  __syncthreads();
  const float tx = txyz[(size_t)t * 3 + 0], ty = txyz[(size_t)t * 3 + 1], tz = txyz[(size_t)t * 3 + 2];
  float best = 3.4e38f; int bi = 0;
  for (int s = 0; s < S; ++s) {
    float dx = tx - sx[s * 3 + 0], dy = ty - sx[s * 3 + 1], dz = tz - sx[s * 3 + 2];
    float d = dx * dx + dy * dy + dz * dz;
    if (d < best) { best = d; bi = s; }               // strict <  == first-index argmin
  }
  const float* fp = sfeat + ((size_t)b * S + bi) * F;
  _Float16* op = outh + (size_t)t * Ds + colOff;
  for (int f = 0; f < F; ++f) op[f] = (_Float16)fp[f];
}

// ---------------- shared MLP GEMM: f16 padded operands, wave tile 16x64 ----------------
// block = 256 = 8 waves covering 128 rows; each wave: 1 A fragment reused by 4 B
// fragments -> 4 independent v_wmma per K-step. Validity of the 4 D-subtiles is
// static per launch (all-4 for D=64/128, single for the D=13 head), so the hot loop
// is specialized branch-free: 10 clause'd b128 loads then 4 back-to-back wmmas.
// A frag: lane half h holds K = h*16 .. h*16+15 sequential.
// B frag: lane half h holds K = h*16 .. h*16+15 sequential.
__global__ __launch_bounds__(256) void mlp_wmma4_k(
    const _Float16* __restrict__ xh,   // [M][Cs], Cs % 32 == 0, zero-padded
    const _Float16* __restrict__ wh,   // [D16][Cs], zero-padded rows+cols
    const float* __restrict__ gamma,   // may be null (head)
    const float* __restrict__ beta,
    float* __restrict__ out,           // [M][D] f32, or null
    _Float16* __restrict__ outh,       // [M][Ds] f16 at colOff, or null
    int M, int Cs, int D, int Ds, int colOff, int do_relu) {
  const int tid  = threadIdx.x;
  const int lane = tid & 31, wave = tid >> 5;
  const int row0 = blockIdx.x * 128;                // M always multiple of 128 here
  const int d0   = blockIdx.y * 64;
  const int h    = lane >> 4;
  const int l15  = lane & 15;
  const int D16  = (D + 15) & ~15;

  const _Float16* ap  = xh + (size_t)(row0 + wave * 16 + l15) * Cs + h * 16;
  const _Float16* bp0 = wh + (size_t)(d0 +  0 + l15) * Cs + h * 16;
  const _Float16* bp1 = wh + (size_t)(d0 + 16 + l15) * Cs + h * 16;
  const _Float16* bp2 = wh + (size_t)(d0 + 32 + l15) * Cs + h * 16;
  const _Float16* bp3 = wh + (size_t)(d0 + 48 + l15) * Cs + h * 16;
  const bool full4 = (d0 + 48) < D16;               // static per launch: 4 tiles or 1

  f32x8 acc0 = {}, acc1 = {}, acc2 = {}, acc3 = {};
  const int ksteps = Cs >> 5;
  if (full4) {
    for (int kt = 0; kt < ksteps; ++kt) {
      // issue all loads first -> one clause group, staged waits before each wmma
      half8 a0 = *(const half8*)(ap);
      half8 a1 = *(const half8*)(ap + 8);
      half8 p0 = *(const half8*)(bp0), p1 = *(const half8*)(bp0 + 8);
      half8 q0 = *(const half8*)(bp1), q1 = *(const half8*)(bp1 + 8);
      half8 r0 = *(const half8*)(bp2), r1 = *(const half8*)(bp2 + 8);
      half8 s0 = *(const half8*)(bp3), s1 = *(const half8*)(bp3 + 8);
      ap += 32; bp0 += 32; bp1 += 32; bp2 += 32; bp3 += 32;
      half16 af, bf0, bf1, bf2, bf3;
      #pragma unroll
      for (int i = 0; i < 8; ++i) {
        af[i]  = a0[i]; af[i + 8]  = a1[i];
        bf0[i] = p0[i]; bf0[i + 8] = p1[i];
        bf1[i] = q0[i]; bf1[i + 8] = q1[i];
        bf2[i] = r0[i]; bf2[i + 8] = r1[i];
        bf3[i] = s0[i]; bf3[i + 8] = s1[i];
      }
      acc0 = __builtin_amdgcn_wmma_f32_16x16x32_f16(false, af, false, bf0, (short)0, acc0, false, false);
      acc1 = __builtin_amdgcn_wmma_f32_16x16x32_f16(false, af, false, bf1, (short)0, acc1, false, false);
      acc2 = __builtin_amdgcn_wmma_f32_16x16x32_f16(false, af, false, bf2, (short)0, acc2, false, false);
      acc3 = __builtin_amdgcn_wmma_f32_16x16x32_f16(false, af, false, bf3, (short)0, acc3, false, false);
    }
  } else {
    for (int kt = 0; kt < ksteps; ++kt) {
      half8 a0 = *(const half8*)(ap);
      half8 a1 = *(const half8*)(ap + 8);
      half8 p0 = *(const half8*)(bp0), p1 = *(const half8*)(bp0 + 8);
      ap += 32; bp0 += 32;
      half16 af, bf0;
      #pragma unroll
      for (int i = 0; i < 8; ++i) {
        af[i]  = a0[i]; af[i + 8]  = a1[i];
        bf0[i] = p0[i]; bf0[i + 8] = p1[i];
      }
      acc0 = __builtin_amdgcn_wmma_f32_16x16x32_f16(false, af, false, bf0, (short)0, acc0, false, false);
    }
  }
  // epilogue: C/D layout — VGPR r: M = r + 8*h, N = lane&15
  auto store_tile = [&](const f32x8& a, int j) {
    const int d = d0 + j * 16 + l15;
    if (d < D) {
      const float gm = gamma ? gamma[d] : 1.f;
      const float bt = beta[d];
      #pragma unroll
      for (int r = 0; r < 8; ++r) {
        int row = row0 + wave * 16 + h * 8 + r;
        float v = a[r] * gm + bt;
        if (do_relu) v = fmaxf(v, 0.f);
        if (out)  out[(size_t)row * D + d] = v;
        if (outh) outh[(size_t)row * Ds + colOff + d] = (_Float16)v;
      }
    }
  };
  store_tile(acc0, 0);
  if (full4) {
    store_tile(acc1, 1);
    store_tile(acc2, 2);
    store_tile(acc3, 3);
  }
}

// =====================================================================================
extern "C" void kernel_launch(void* const* d_in, const int* in_sizes, int n_in,
                              void* d_out, int out_size, void* d_ws, size_t ws_size,
                              hipStream_t stream) {
  (void)in_sizes; (void)n_in; (void)out_size; (void)ws_size;
  const float* points  = (const float*)d_in[0];
  const float* w0      = (const float*)d_in[1];
  const float* g0      = (const float*)d_in[2];
  const float* b0      = (const float*)d_in[3];
  const float* w512    = (const float*)d_in[4];
  const float* g512    = (const float*)d_in[5];
  const float* b512    = (const float*)d_in[6];
  const float* w128    = (const float*)d_in[7];
  const float* g128    = (const float*)d_in[8];
  const float* b128    = (const float*)d_in[9];
  const float* wpos512 = (const float*)d_in[10];
  const float* gpos512 = (const float*)d_in[11];
  const float* bpos512 = (const float*)d_in[12];
  const float* wposN   = (const float*)d_in[13];
  const float* gposN   = (const float*)d_in[14];
  const float* bposN   = (const float*)d_in[15];
  const float* wup1    = (const float*)d_in[16];
  const float* gup1    = (const float*)d_in[17];
  const float* bup1    = (const float*)d_in[18];
  const float* wup2    = (const float*)d_in[19];
  const float* gup2    = (const float*)d_in[20];
  const float* bup2    = (const float*)d_in[21];
  const float* whead   = (const float*)d_in[22];
  const float* bhead   = (const float*)d_in[23];

  char* ws = (char*)d_ws;
  size_t off = 0;
  auto allocB = [&](size_t bytes) -> void* {
    void* p = (void*)(ws + off);
    off += (bytes + 255) & ~(size_t)255;
    return p;
  };
  // f32 intermediates
  float* xyzn    = (float*)allocB((size_t)65536 * 3 * 4);
  float* feat0   = (float*)allocB((size_t)65536 * 64 * 4);
  float* xyz512  = (float*)allocB((size_t)4096 * 3 * 4);
  float* feat512 = (float*)allocB((size_t)4096 * 64 * 4);
  float* xyz128  = (float*)allocB((size_t)1024 * 3 * 4);
  float* feat128 = (float*)allocB((size_t)1024 * 128 * 4);
  float* fuse512 = (float*)allocB((size_t)4096 * 128 * 4);
  int*   idx512  = (int*)allocB(8 * 512 * 4);
  int*   idx128  = (int*)allocB(8 * 128 * 4);
  // f16 padded GEMM operands
  _Float16* xp0_h     = (_Float16*)allocB((size_t)65536 * 32 * 2);
  _Float16* xyzn_h    = (_Float16*)allocB((size_t)65536 * 32 * 2);
  _Float16* xyz512_h  = (_Float16*)allocB((size_t)4096 * 32 * 2);
  _Float16* cat512_h  = (_Float16*)allocB((size_t)4096 * 96 * 2);
  _Float16* cat128_h  = (_Float16*)allocB((size_t)1024 * 96 * 2);
  _Float16* cat320_h  = (_Float16*)allocB((size_t)4096 * 320 * 2);
  _Float16* cat256_h  = (_Float16*)allocB((size_t)65536 * 256 * 2);
  _Float16* fuseN_h   = (_Float16*)allocB((size_t)65536 * 128 * 2);
  // f16 packed weights
  _Float16* w0h      = (_Float16*)allocB((size_t)64 * 32 * 2);
  _Float16* w512h    = (_Float16*)allocB((size_t)64 * 96 * 2);
  _Float16* w128h    = (_Float16*)allocB((size_t)128 * 96 * 2);
  _Float16* wpos512h = (_Float16*)allocB((size_t)128 * 32 * 2);
  _Float16* wposNh   = (_Float16*)allocB((size_t)64 * 32 * 2);
  _Float16* wup1h    = (_Float16*)allocB((size_t)128 * 320 * 2);
  _Float16* wup2h    = (_Float16*)allocB((size_t)128 * 256 * 2);
  _Float16* wheadh   = (_Float16*)allocB((size_t)16 * 128 * 2);
  float* logits = (float*)d_out;

  auto g1 = [](int total) { return dim3((total + 255) / 256); };

  // --- zero padded cols of concat buffers that aren't fully overwritten ---
  fill0_k<<<g1(4096 * 96 / 2), 256, 0, stream>>>((unsigned int*)cat512_h, 4096 * 96 / 2);
  fill0_k<<<g1(1024 * 96 / 2), 256, 0, stream>>>((unsigned int*)cat128_h, 1024 * 96 / 2);

  // --- pack weights to f16 [D16][Cpad] ---
  pack_f16_k<<<g1(64 * 32),   256, 0, stream>>>(w0h,      w0,      64,  9,   32,  64 * 32);
  pack_f16_k<<<g1(64 * 96),   256, 0, stream>>>(w512h,    w512,    64,  67,  96,  64 * 96);
  pack_f16_k<<<g1(128 * 96),  256, 0, stream>>>(w128h,    w128,    128, 67,  96,  128 * 96);
  pack_f16_k<<<g1(128 * 32),  256, 0, stream>>>(wpos512h, wpos512, 128, 3,   32,  128 * 32);
  pack_f16_k<<<g1(64 * 32),   256, 0, stream>>>(wposNh,   wposN,   64,  3,   32,  64 * 32);
  pack_f16_k<<<g1(128 * 320), 256, 0, stream>>>(wup1h,    wup1,    128, 320, 320, 128 * 320);
  pack_f16_k<<<g1(128 * 256), 256, 0, stream>>>(wup2h,    wup2,    128, 256, 256, 128 * 256);
  pack_f16_k<<<g1(16 * 128),  256, 0, stream>>>(wheadh,   whead,   13,  128, 128, 16 * 128);

  // --- stage 1: normalize, feat0 (writes f32 + f16 slice of cat256) ---
  normalize_k<<<8, 256, 0, stream>>>(points, xyzn);
  pack_f16_k<<<g1(65536 * 32), 256, 0, stream>>>(xp0_h, points, 65536, 9, 32, 65536 * 32);
  mlp_wmma4_k<<<dim3(512, 1), 256, 0, stream>>>(xp0_h, w0h, g0, b0,
                                                feat0, cat256_h, 65536, 32, 64, 256, 128, 1);
  pack_f16_k<<<g1(65536 * 32), 256, 0, stream>>>(xyzn_h, xyzn, 65536, 3, 32, 65536 * 32);

  // --- FPS level 1, gathers into cat512_h, MLP512 (f32 + f16 slice of cat320) ---
  fps_k<<<8, 1024, 0, stream>>>(xyzn, idx512, 8192, 512);
  gather_f32_k<<<g1(8 * 512 * 3), 256, 0, stream>>>(xyz512, xyzn, idx512, 8192, 512, 3, 8 * 512 * 3);
  gather_f16s_k<<<g1(8 * 512 * 3), 256, 0, stream>>>(cat512_h, 96, 0, xyzn, idx512, 8192, 512, 3, 8 * 512 * 3);
  gather_f16s_k<<<g1(8 * 512 * 64), 256, 0, stream>>>(cat512_h, 96, 3, feat0, idx512, 8192, 512, 64, 8 * 512 * 64);
  mlp_wmma4_k<<<dim3(32, 1), 256, 0, stream>>>(cat512_h, w512h, g512, b512,
                                               feat512, cat320_h, 4096, 96, 64, 320, 0, 1);
  pack_f16_k<<<g1(4096 * 32), 256, 0, stream>>>(xyz512_h, xyz512, 4096, 3, 32, 4096 * 32);

  // --- FPS level 2, gathers into cat128_h, MLP128 ---
  fps_k<<<8, 1024, 0, stream>>>(xyz512, idx128, 512, 128);
  gather_f32_k<<<g1(8 * 128 * 3), 256, 0, stream>>>(xyz128, xyz512, idx128, 512, 128, 3, 8 * 128 * 3);
  gather_f16s_k<<<g1(8 * 128 * 3), 256, 0, stream>>>(cat128_h, 96, 0, xyz512, idx128, 512, 128, 3, 8 * 128 * 3);
  gather_f16s_k<<<g1(8 * 128 * 64), 256, 0, stream>>>(cat128_h, 96, 3, feat512, idx128, 512, 128, 64, 8 * 128 * 64);
  mlp_wmma4_k<<<dim3(8, 2), 256, 0, stream>>>(cat128_h, w128h, g128, b128,
                                              feat128, (_Float16*)nullptr, 1024, 96, 128, 0, 0, 1);

  // --- up 128->512 and pos512 directly into cat320_h; fuse512 ---
  nn_interp_k<<<(8 * 512) / 256, 256, 0, stream>>>(xyz512, xyz128, feat128, cat320_h, 512, 128, 128, 320, 64);
  mlp_wmma4_k<<<dim3(32, 2), 256, 0, stream>>>(xyz512_h, wpos512h, gpos512, bpos512,
                                               (float*)nullptr, cat320_h, 4096, 32, 128, 320, 192, 1);
  mlp_wmma4_k<<<dim3(32, 2), 256, 0, stream>>>(cat320_h, wup1h, gup1, bup1,
                                               fuse512, (_Float16*)nullptr, 4096, 320, 128, 0, 0, 1);

  // --- up 512->N and posN directly into cat256_h; fuseN (f16 only); head ---
  nn_interp_k<<<(8 * 8192) / 256, 256, 0, stream>>>(xyzn, xyz512, fuse512, cat256_h, 8192, 512, 128, 256, 0);
  mlp_wmma4_k<<<dim3(512, 1), 256, 0, stream>>>(xyzn_h, wposNh, gposN, bposN,
                                                (float*)nullptr, cat256_h, 65536, 32, 64, 256, 192, 1);
  mlp_wmma4_k<<<dim3(512, 2), 256, 0, stream>>>(cat256_h, wup2h, gup2, bup2,
                                                (float*)nullptr, fuseN_h, 65536, 256, 128, 128, 0, 1);
  mlp_wmma4_k<<<dim3(512, 1), 256, 0, stream>>>(fuseN_h, wheadh, (const float*)nullptr, bhead,
                                                logits, (_Float16*)nullptr, 65536, 128, 13, 0, 0, 0);
}
